// stawnet_17351667876077
// MI455X (gfx1250) — matrix operand order repaired and checked by reading, not compiled, not run on hardware
//
#include <hip/hip_runtime.h>

typedef __attribute__((ext_vector_type(16))) _Float16 v16h;
typedef __attribute__((ext_vector_type(8)))  float    v8f;

#define NB 64
#define NN 325
#define NPAD 352   // multiple of 32 >= NN

// ---------------------------------------------------------------------------
// start conv: (B,2,N,13) -> (B,32,N,13)
// ---------------------------------------------------------------------------
__global__ __launch_bounds__(256) void st_start(
    const float* __restrict__ xin, const float* __restrict__ W,
    const float* __restrict__ bias, float* __restrict__ xout, int total)
{
  __shared__ float w[64], bb[32];
  int tid = threadIdx.x;
  if (tid < 64) w[tid] = W[tid];
  if (tid < 32) bb[tid] = bias[tid];
  __syncthreads();
  int idx = blockIdx.x * 256 + tid;
  if (idx >= total) return;
  int t = idx % 13; int bn = idx / 13; int n = bn % NN; int b = bn / NN;
  float a0 = xin[(((size_t)b*2 + 0)*NN + n)*13 + t];
  float a1 = xin[(((size_t)b*2 + 1)*NN + n)*13 + t];
  #pragma unroll 4
  for (int o = 0; o < 32; ++o)
    xout[(((size_t)b*32 + o)*NN + n)*13 + t] = w[o*2]*a0 + w[o*2+1]*a1 + bb[o];
}

// ---------------------------------------------------------------------------
// gated dilated conv: x(B,32,N,T) -> gated f32 (B,32,N,To) + gatedT f16 (B,To,N,32)
// ---------------------------------------------------------------------------
__global__ __launch_bounds__(256) void st_gate(
    const float* __restrict__ x,
    const float* __restrict__ fW, const float* __restrict__ fb,
    const float* __restrict__ gW, const float* __restrict__ gb,
    float* __restrict__ gated, _Float16* __restrict__ gT,
    int T, int To, int dil, int layer, int total)
{
  __shared__ float wf[2048], wg[2048], bf[32], bg[32];
  int tid = threadIdx.x;
  for (int i = tid; i < 2048; i += 256) {
    wf[i] = fW[layer*2048 + i];
    wg[i] = gW[layer*2048 + i];
  }
  if (tid < 32) { bf[tid] = fb[layer*32 + tid]; bg[tid] = gb[layer*32 + tid]; }
  __syncthreads();
  int idx = blockIdx.x * 256 + tid;
  if (idx >= total) return;          // total = B*N*To
  int t = idx % To; int bn = idx / To; int n = bn % NN; int b = bn / NN;
  float xl[32], xr[32];
  #pragma unroll 4
  for (int c = 0; c < 32; ++c) {
    size_t base = (((size_t)b*32 + c)*NN + n)*T + t;
    xl[c] = x[base];
    xr[c] = x[base + dil];
  }
  for (int o = 0; o < 32; ++o) {
    float f = bf[o], g = bg[o];
    #pragma unroll 8
    for (int c = 0; c < 32; ++c) {
      f += wf[o*64 + 2*c] * xl[c] + wf[o*64 + 2*c + 1] * xr[c];
      g += wg[o*64 + 2*c] * xl[c] + wg[o*64 + 2*c + 1] * xr[c];
    }
    float v = tanhf(f) * (1.f / (1.f + __expf(-g)));
    gated[(((size_t)b*32 + o)*NN + n)*To + t] = v;
    gT[(((size_t)b*To + t)*NN + n)*32 + o] = (_Float16)v;
  }
}

// ---------------------------------------------------------------------------
// q/k projections on concat(x, embedding): -> qT,kT f16 (B,To,N,16)
// ---------------------------------------------------------------------------
__global__ __launch_bounds__(256) void st_qk(
    const _Float16* __restrict__ gT, const float* __restrict__ emb,
    const float* __restrict__ qW, const float* __restrict__ qb,
    const float* __restrict__ kW, const float* __restrict__ kb,
    _Float16* __restrict__ qT, _Float16* __restrict__ kT,
    int layer, int total)
{
  __shared__ float wq[768], wk[768], bq[16], bk[16];
  int tid = threadIdx.x;
  for (int i = tid; i < 768; i += 256) {
    wq[i] = qW[layer*768 + i];
    wk[i] = kW[layer*768 + i];
  }
  if (tid < 16) { bq[tid] = qb[layer*16 + tid]; bk[tid] = kb[layer*16 + tid]; }
  __syncthreads();
  int idx = blockIdx.x * 256 + tid;
  if (idx >= total) return;          // total = B*To*N, idx = (b*To+t)*N + n
  int n = idx % NN;
  float xv[48];
  const _Float16* gp = gT + (size_t)idx * 32;
  #pragma unroll 8
  for (int c = 0; c < 32; ++c) xv[c] = (float)gp[c];
  #pragma unroll 4
  for (int e = 0; e < 16; ++e) xv[32 + e] = emb[e*NN + n];
  for (int d = 0; d < 16; ++d) {
    float q = bq[d], k = bk[d];
    #pragma unroll 8
    for (int c = 0; c < 48; ++c) { q += wq[d*48 + c]*xv[c]; k += wk[d*48 + c]*xv[c]; }
    qT[(size_t)idx*16 + d] = (_Float16)q;
    kT[(size_t)idx*16 + d] = (_Float16)k;
  }
}

// ---------------------------------------------------------------------------
// softmax row stats: rowmax/rowsum (B,To,N).  grid (2, To, B)
// ---------------------------------------------------------------------------
__global__ __launch_bounds__(256) void st_stats(
    const _Float16* __restrict__ qT, const _Float16* __restrict__ kT,
    float* __restrict__ rowmax, float* __restrict__ rowsum, int T)
{
  __shared__ _Float16 ks[NN * 16];
  int tid = threadIdx.x;
  int t = blockIdx.y, b = blockIdx.z;
  size_t btn = ((size_t)b*T + t) * NN;
  for (int i = tid; i < NN*16; i += 256) ks[i] = kT[btn*16 + i];
  __syncthreads();
  int n = blockIdx.x * 256 + tid;
  if (n >= NN) return;
  float qv[16];
  #pragma unroll 4
  for (int d = 0; d < 16; ++d) qv[d] = (float)qT[(btn + n)*16 + d];
  float mr = -3.0e38f, lr = 0.f;
  for (int m = 0; m < NN; ++m) {
    float s = 0.f;
    #pragma unroll
    for (int d = 0; d < 16; ++d) s += qv[d] * (float)ks[m*16 + d];
    s *= 0.25f;                      // 1/sqrt(D_ATT)
    float nm = fmaxf(mr, s);
    lr = lr * __expf(mr - nm) + __expf(s - nm);
    mr = nm;
  }
  rowmax[btn + n] = mr;
  rowsum[btn + n] = lr;
}

// ---------------------------------------------------------------------------
// fused WMMA attention:  s = q^T k (WMMA), p = exp(s-rm)/rs, xa = x * p (WMMA)
// grid (6, To, B), 128 threads (4 waves), wave w owns m-tile m0 = bx*64 + w*16
// ---------------------------------------------------------------------------
__global__ __launch_bounds__(128) void st_att(
    const _Float16* __restrict__ qT, const _Float16* __restrict__ kT,
    const _Float16* __restrict__ gT, const float* __restrict__ rowmax,
    const float* __restrict__ rowsum, float* __restrict__ xa, int T)
{
  __shared__ _Float16 xc[32][NPAD];     // x as [c][n], f16
  __shared__ _Float16 qs[NPAD][16];     // q as [n][d], f16
  __shared__ float rmax[NPAD];
  __shared__ float rinv[NPAD];
  __shared__ _Float16 Ps[4][32][16];    // per-wave P staging (K=n_local, N=m_local)

  const int t = blockIdx.y;
  const int b = blockIdx.z;
  const int tid = threadIdx.x;
  const int wave = tid >> 5;
  const int lane = tid & 31;
  const int hw = lane >> 4;             // which 16-lane half
  const int l16 = lane & 15;

  const size_t btn = ((size_t)b*T + t) * NN;

  for (int i = tid; i < NPAD*32; i += 128) {
    int n = i >> 5, c = i & 31;
    _Float16 v = (_Float16)0.f;
    if (n < NN) v = gT[(btn + n)*32 + c];
    xc[c][n] = v;
  }
  for (int i = tid; i < NPAD*16; i += 128) {
    int n = i >> 4, d = i & 15;
    _Float16 v = (_Float16)0.f;
    if (n < NN) v = qT[(btn + n)*16 + d];
    qs[n][d] = v;
  }
  for (int n = tid; n < NPAD; n += 128) {
    rmax[n] = (n < NN) ? rowmax[btn + n] : 0.f;
    rinv[n] = (n < NN) ? (1.f / rowsum[btn + n]) : 0.f;
  }
  __syncthreads();

  const int m0 = blockIdx.x * 64 + wave * 16;

  // B operand of s-WMMA: k as (d x m), K = 16*hw + e -> d ; d>=16 is zero pad
  v16h kb;
  #pragma unroll
  for (int e = 0; e < 16; ++e) kb[e] = (_Float16)0.f;
  if (hw == 0) {
    int m = m0 + l16;
    if (m < NN) {
      const _Float16* kp = kT + (btn + m)*16;
      #pragma unroll
      for (int e = 0; e < 16; ++e) kb[e] = kp[e];
    }
  }

  v8f acc0 = {};   // xa rows c=0..15
  v8f acc1 = {};   // xa rows c=16..31

  for (int n0 = 0; n0 < NPAD; n0 += 32) {
    #pragma unroll
    for (int tt = 0; tt < 2; ++tt) {
      const int ntile = n0 + tt*16;
      // A operand: q^T rows n (M), K=d padded to 32
      v16h qa;
      int n = ntile + l16;
      #pragma unroll
      for (int e = 0; e < 16; ++e) {
        int K = 16*hw + e;
        qa[e] = (K < 16) ? qs[n][K] : (_Float16)0.f;
      }
      v8f s = {};
      s = __builtin_amdgcn_wmma_f32_16x16x32_f16(false, qa, false, kb,
                                                 (short)0, s, false, false);
      // p = exp(s/4 - rowmax)/rowsum ; zero pad rows ; stage into LDS
      #pragma unroll
      for (int r = 0; r < 8; ++r) {
        int row = r + 8*hw;
        int nn = ntile + row;
        float p = 0.f;
        if (nn < NN) p = __expf(s[r]*0.25f - rmax[nn]) * rinv[nn];
        Ps[wave][tt*16 + row][l16] = (_Float16)p;
      }
    }
    // intra-wave LDS write->read ordering + compiler barrier
    asm volatile("s_wait_dscnt 0" ::: "memory");
    // B operand from P: K = 16*hw + e (n_local), N = l16 (m_local)
    v16h pb;
    #pragma unroll
    for (int e = 0; e < 16; ++e) pb[e] = Ps[wave][16*hw + e][l16];
    // A operands from x: rows c, K = n_local
    #pragma unroll
    for (int ct = 0; ct < 2; ++ct) {
      v16h xav;
      int c = ct*16 + l16;
      #pragma unroll
      for (int e = 0; e < 16; ++e) {
        int K = 16*hw + e;
        xav[e] = xc[c][n0 + K];
      }
      if (ct == 0)
        acc0 = __builtin_amdgcn_wmma_f32_16x16x32_f16(false, xav, false, pb,
                                                      (short)0, acc0, false, false);
      else
        acc1 = __builtin_amdgcn_wmma_f32_16x16x32_f16(false, xav, false, pb,
                                                      (short)0, acc1, false, false);
    }
  }

  // store: C layout lane holds (M = r + 8*hw, N = l16); xa (B,32,N,To)
  int m = m0 + l16;
  if (m < NN) {
    #pragma unroll
    for (int r = 0; r < 8; ++r) {
      int c0 = r + 8*hw;
      xa[(((size_t)b*32 + c0     )*NN + m)*T + t] = acc0[r];
      xa[(((size_t)b*32 + c0 + 16)*NN + m)*T + t] = acc1[r];
    }
  }
}

// ---------------------------------------------------------------------------
// skip conv, last time index only (only slice that survives accumulation)
// grid (N, B), 256 threads (s index)
// ---------------------------------------------------------------------------
__global__ __launch_bounds__(256) void st_skip(
    const float* __restrict__ gated, const float* __restrict__ sW,
    const float* __restrict__ sb, float* __restrict__ skip,
    int To, int layer)
{
  __shared__ float xl[32];
  int n = blockIdx.x, b = blockIdx.y, tid = threadIdx.x;
  if (tid < 32) xl[tid] = gated[(((size_t)b*32 + tid)*NN + n)*To + (To - 1)];
  __syncthreads();
  float acc = sb[layer*256 + tid];
  const float* w = sW + ((size_t)layer*256 + tid)*32;
  #pragma unroll 8
  for (int c = 0; c < 32; ++c) acc += w[c]*xl[c];
  skip[((size_t)b*256 + tid)*NN + n] += acc;
}

// ---------------------------------------------------------------------------
// mlp on concat(gated, xa) + residual + batchnorm -> x_next (B,32,N,To)
// ---------------------------------------------------------------------------
__global__ __launch_bounds__(256) void st_mlp(
    const float* __restrict__ gated, const float* __restrict__ xa,
    const float* __restrict__ xprev,
    const float* __restrict__ mW, const float* __restrict__ mb,
    const float* __restrict__ gam, const float* __restrict__ bet,
    const float* __restrict__ mu, const float* __restrict__ var,
    float* __restrict__ xnext, int T, int To, int dil, int layer, int total)
{
  __shared__ float w[2048], bias[32], sg[32], sb2[32], smu[32], siv[32];
  int tid = threadIdx.x;
  for (int i = tid; i < 2048; i += 256) w[i] = mW[layer*2048 + i];
  if (tid < 32) {
    bias[tid] = mb[layer*32 + tid];
    sg[tid]   = gam[layer*32 + tid];
    sb2[tid]  = bet[layer*32 + tid];
    smu[tid]  = mu[layer*32 + tid];
    siv[tid]  = rsqrtf(var[layer*32 + tid] + 1e-5f);
  }
  __syncthreads();
  int idx = blockIdx.x * 256 + tid;
  if (idx >= total) return;          // total = B*N*To
  int t = idx % To; int bn = idx / To; int n = bn % NN; int b = bn / NN;
  float xg[32], xv[32];
  #pragma unroll 4
  for (int c = 0; c < 32; ++c) {
    size_t base = (((size_t)b*32 + c)*NN + n)*To + t;
    xg[c] = gated[base];
    xv[c] = xa[base];
  }
  for (int o = 0; o < 32; ++o) {
    float v = bias[o];
    #pragma unroll 8
    for (int c = 0; c < 32; ++c) v += w[o*64 + c]*xg[c] + w[o*64 + 32 + c]*xv[c];
    v += xprev[(((size_t)b*32 + o)*NN + n)*T + t + dil];
    v = (v - smu[o]) * siv[o] * sg[o] + sb2[o];
    xnext[(((size_t)b*32 + o)*NN + n)*To + t] = v;
  }
}

// ---------------------------------------------------------------------------
// head: relu(skip) -> relu(W1) -> W2 ; grid (N, B)
// ---------------------------------------------------------------------------
__global__ __launch_bounds__(256) void st_end(
    const float* __restrict__ skip, const float* __restrict__ W1,
    const float* __restrict__ b1, const float* __restrict__ W2,
    const float* __restrict__ b2, float* __restrict__ out)
{
  __shared__ float sk[256];
  __shared__ float h[512];
  int n = blockIdx.x, b = blockIdx.y, tid = threadIdx.x;
  sk[tid] = fmaxf(skip[((size_t)b*256 + tid)*NN + n], 0.f);
  __syncthreads();
  for (int e = tid; e < 512; e += 256) {
    float a = b1[e];
    const float* w = W1 + (size_t)e*256;
    #pragma unroll 8
    for (int s = 0; s < 256; ++s) a += w[s]*sk[s];
    h[e] = fmaxf(a, 0.f);
  }
  __syncthreads();
  if (tid < 12) {
    float a = b2[tid];
    const float* w = W2 + tid*512;
    #pragma unroll 8
    for (int e = 0; e < 512; ++e) a += w[e]*h[e];
    out[((size_t)b*12 + tid)*NN + n] = a;
  }
}

__global__ void st_zero(float* __restrict__ p, int n) {
  int i = blockIdx.x*256 + threadIdx.x;
  if (i < n) p[i] = 0.f;
}

// ---------------------------------------------------------------------------
extern "C" void kernel_launch(void* const* d_in, const int* in_sizes, int n_in,
                              void* d_out, int out_size, void* d_ws, size_t ws_size,
                              hipStream_t stream) {
  const float* x_in   = (const float*)d_in[0];
  const float* emb    = (const float*)d_in[1];
  const float* startW = (const float*)d_in[2];
  const float* startB = (const float*)d_in[3];
  const float* fW     = (const float*)d_in[4];
  const float* fB     = (const float*)d_in[5];
  const float* gW     = (const float*)d_in[6];
  const float* gB     = (const float*)d_in[7];
  const float* skW    = (const float*)d_in[8];
  const float* skB    = (const float*)d_in[9];
  const float* qW     = (const float*)d_in[10];
  const float* qB     = (const float*)d_in[11];
  const float* kW     = (const float*)d_in[12];
  const float* kB     = (const float*)d_in[13];
  const float* mW     = (const float*)d_in[14];
  const float* mB     = (const float*)d_in[15];
  const float* bnG    = (const float*)d_in[16];
  const float* bnB    = (const float*)d_in[17];
  const float* bnM    = (const float*)d_in[18];
  const float* bnV    = (const float*)d_in[19];
  const float* e1W    = (const float*)d_in[20];
  const float* e1B    = (const float*)d_in[21];
  const float* e2W    = (const float*)d_in[22];
  const float* e2B    = (const float*)d_in[23];
  float* out = (float*)d_out;

  const int B = NB, N = NN;
  char* ws = (char*)d_ws;
  size_t off = 0;
  auto take = [&](size_t bytes) -> char* {
    char* p = ws + off;
    off += (bytes + 255) & ~(size_t)255;
    return p;
  };
  float*    x0    = (float*)take((size_t)B*32*N*13*4);
  float*    x1    = (float*)take((size_t)B*32*N*13*4);
  float*    gated = (float*)take((size_t)B*32*N*12*4);
  float*    xa    = (float*)take((size_t)B*32*N*12*4);
  _Float16* gT    = (_Float16*)take((size_t)B*12*N*32*2);
  _Float16* qT    = (_Float16*)take((size_t)B*12*N*16*2);
  _Float16* kT    = (_Float16*)take((size_t)B*12*N*16*2);
  float*    rmax  = (float*)take((size_t)B*12*N*4);
  float*    rsum  = (float*)take((size_t)B*12*N*4);
  float*    skip  = (float*)take((size_t)B*256*N*4);

  // start conv
  {
    int total = B*N*13;
    st_start<<<(total + 255)/256, 256, 0, stream>>>(x_in, startW, startB, x0, total);
  }
  // zero skip accumulator
  {
    int nz = B*256*N;
    st_zero<<<(nz + 255)/256, 256, 0, stream>>>(skip, nz);
  }

  static const int dils[8] = {1, 2, 1, 2, 1, 2, 1, 2};
  int T = 13;
  float* xc = x0;
  float* xn = x1;
  for (int i = 0; i < 8; ++i) {
    const int dil = dils[i];
    const int To = T - dil;

    {
      int total = B*N*To;
      st_gate<<<(total + 255)/256, 256, 0, stream>>>(
          xc, fW, fB, gW, gB, gated, gT, T, To, dil, i, total);
    }
    {
      int total = B*To*N;
      st_qk<<<(total + 255)/256, 256, 0, stream>>>(
          gT, emb, qW, qB, kW, kB, qT, kT, i, total);
    }
    {
      dim3 grid(2, To, B);
      st_stats<<<grid, 256, 0, stream>>>(qT, kT, rmax, rsum, To);
    }
    {
      dim3 grid((N + 63)/64, To, B);   // 6 m-blocks of 64 (4 waves x 16)
      st_att<<<grid, 128, 0, stream>>>(qT, kT, gT, rmax, rsum, xa, To);
    }
    {
      dim3 grid(N, B);
      st_skip<<<grid, 256, 0, stream>>>(gated, skW, skB, skip, To, i);
    }
    {
      int total = B*N*To;
      st_mlp<<<(total + 255)/256, 256, 0, stream>>>(
          gated, xa, xc, mW, mB, bnG, bnB, bnM, bnV, xn, T, To, dil, i, total);
    }
    T = To;
    float* tmp = xc; xc = xn; xn = tmp;
  }

  {
    dim3 grid(N, B);
    st_end<<<grid, 256, 0, stream>>>(skip, e1W, e1B, e2W, e2B, out);
  }
}
